// SelfAttention1D_52536039964800
// MI455X (gfx1250) — matrix operand order repaired and checked by reading, not compiled, not run on hardware
//
#include <hip/hip_runtime.h>

// SelfAttention1D for MI455X (gfx1250): all GEMMs via v_wmma_f32_16x16x32_f16,
// f32 accumulation, fp32 softmax. B=32, C=512, C8=64, L=1024.
// Workspace layout (f16): Qh[32][1024][64], Kh[32][1024][64], Vh[32][512][1024] = 40MB.

typedef __attribute__((ext_vector_type(16))) _Float16 v16h;
typedef __attribute__((ext_vector_type(8)))  _Float16 v8h;
typedef __attribute__((ext_vector_type(8)))  float    v8f;

__device__ __forceinline__ v8f wmma_f16(v16h a, v16h b, v8f c) {
  // (neg_a, A, neg_b, B, c_mod, C, reuse_a, reuse_b)
  return __builtin_amdgcn_wmma_f32_16x16x32_f16(false, a, false, b, (short)0, c,
                                                false, false);
}

__device__ __forceinline__ v16h pack16(v8h lo, v8h hi) {
  v16h r;
#pragma unroll
  for (int i = 0; i < 8; ++i) { r[i] = lo[i]; r[8 + i] = hi[i]; }
  return r;
}

// A fragment (16x32 f16) from row-major [16][stride] halves.
// ISA layout: lane(g,m): halves 0..7 <- K=8g..8g+7 ; halves 8..15 <- K=16+8g..+7
__device__ __forceinline__ v16h load_a_f16(const _Float16* base, int stride) {
  const int lane = threadIdx.x & 31, g = lane >> 4, m = lane & 15;
  const _Float16* p = base + m * stride + 8 * g;
  return pack16(*(const v8h*)p, *(const v8h*)(p + 16));
}

__device__ __forceinline__ v16h cvt16(const float* p0, const float* p1) {
  float f[16];
  *(float4*)(f + 0)  = ((const float4*)p0)[0];
  *(float4*)(f + 4)  = ((const float4*)p0)[1];
  *(float4*)(f + 8)  = ((const float4*)p1)[0];
  *(float4*)(f + 12) = ((const float4*)p1)[1];
  v16h r;
#pragma unroll
  for (int i = 0; i < 16; ++i) r[i] = (_Float16)f[i];
  return r;
}

// A fragment from row-major fp32 [16][stride] (converted to f16).
__device__ __forceinline__ v16h load_a_f32(const float* base, int stride) {
  const int lane = threadIdx.x & 31, g = lane >> 4, m = lane & 15;
  const float* p = base + m * stride + 8 * g;
  return cvt16(p, p + 16);
}

// B fragment (32x16 f16) from column-major storage [N][K] (stride in halves).
// ISA layout: lane(g,n) holds column n, halves 0..7 <- K=8g..8g+7 ; halves 8..15 <- K=16+8g..+7
__device__ __forceinline__ v16h load_b_f16(const _Float16* base, int stride) {
  const int lane = threadIdx.x & 31, g = lane >> 4, n = lane & 15;
  const _Float16* p = base + n * stride + 8 * g;
  return pack16(*(const v8h*)p, *(const v8h*)(p + 16));
}

// B fragment from column-major fp32 [N][K] storage (converted to f16).
__device__ __forceinline__ v16h load_b_f32(const float* base, int stride) {
  const int lane = threadIdx.x & 31, g = lane >> 4, n = lane & 15;
  const float* p = base + n * stride + 8 * g;
  return cvt16(p, p + 16);
}

// ---------------------------------------------------------------------------
// Kernel 1: fused QKV projection. One WG = (batch b, 32 l-positions).
// Q[l][d] = sum_c Xt[l][c] * Wq[d][c]   (A = Xt rows, B = Wq rows)
// V[c][l] = sum_c' Wv[c][c'] * Xt[l][c'] (A = Wv rows, B = Xt as [N=l][K=c])
// ---------------------------------------------------------------------------
__global__ __launch_bounds__(256) void proj_kernel(
    const float* __restrict__ x,  const float* __restrict__ Wq,
    const float* __restrict__ bq, const float* __restrict__ Wk,
    const float* __restrict__ bk, const float* __restrict__ Wv,
    const float* __restrict__ bv, _Float16* __restrict__ Qh,
    _Float16* __restrict__ Kh,    _Float16* __restrict__ Vh) {
  const int b  = blockIdx.x >> 5;
  const int l0 = (blockIdx.x & 31) * 32;
  const int tid = threadIdx.x;
  const int w = tid >> 5, lane = tid & 31, g = lane >> 4, n = lane & 15;

  __shared__ _Float16 Xt[32 * 40];   // [l][c-chunk] f16, padded stride 40
  __shared__ float bvS[512];
  bvS[tid] = bv[tid];
  bvS[tid + 256] = bv[tid + 256];

  v8f accQK[2]   = {};   // this wave's 2 of 16 Q/K tiles
  v8f accV[4][2] = {};   // c-tiles 4w..4w+3  x  l-tiles 0..1
  const int t0 = 2 * w;

  for (int kk = 0; kk < 16; ++kk) {  // K loop over c in steps of 32
    const int c0 = kk * 32;
    __syncthreads();
    {  // stage x[b][c0..c0+32][l0..l0+32] transposed into LDS as f16
      const int cc = tid >> 3, ll = (tid & 7) * 4;
      float4 xv = *(const float4*)(x + ((size_t)(b * 512 + c0 + cc)) * 1024 + l0 + ll);
      Xt[(ll + 0) * 40 + cc] = (_Float16)xv.x;
      Xt[(ll + 1) * 40 + cc] = (_Float16)xv.y;
      Xt[(ll + 2) * 40 + cc] = (_Float16)xv.z;
      Xt[(ll + 3) * 40 + cc] = (_Float16)xv.w;
    }
    __syncthreads();

    // --- Q/K tiles (2 per wave) ---
#pragma unroll
    for (int t = 0; t < 2; ++t) {
      const int id = t0 + t, qk = id >> 3, r = id & 7, lt = r >> 2, dt = r & 3;
      v16h a = load_a_f16(Xt + lt * 16 * 40, 40);
      const float* W = qk ? Wk : Wq;                     // [C8][C] row-major
      v16h bw = load_b_f32(W + (dt * 16) * 512 + c0, 512);
      accQK[t] = wmma_f16(a, bw, accQK[t]);
    }
    // --- V tiles (8 per wave) ---
    v16h bx0 = load_b_f16(Xt, 40);
    v16h bx1 = load_b_f16(Xt + 16 * 40, 40);
#pragma unroll
    for (int ct = 0; ct < 4; ++ct) {
      v16h av = load_a_f32(Wv + (size_t)(w * 64 + ct * 16) * 512 + c0, 512);
      accV[ct][0] = wmma_f16(av, bx0, accV[ct][0]);
      accV[ct][1] = wmma_f16(av, bx1, accV[ct][1]);
    }
  }

  // epilogue: add biases, convert to f16, store
#pragma unroll
  for (int t = 0; t < 2; ++t) {
    const int id = t0 + t, qk = id >> 3, r = id & 7, lt = r >> 2, dt = r & 3;
    const int d = dt * 16 + n;
    const float bias = qk ? bk[d] : bq[d];
    _Float16* O = qk ? Kh : Qh;
#pragma unroll
    for (int i = 0; i < 8; ++i) {
      const int l = l0 + lt * 16 + i + 8 * g;
      O[(b * 1024 + l) * 64 + d] = (_Float16)(accQK[t][i] + bias);
    }
  }
#pragma unroll
  for (int ct = 0; ct < 4; ++ct)
#pragma unroll
    for (int lt = 0; lt < 2; ++lt)
#pragma unroll
      for (int i = 0; i < 8; ++i) {
        const int c = w * 64 + ct * 16 + i + 8 * g;
        const int l = l0 + lt * 16 + n;
        Vh[(b * 512 + c) * 1024 + l] = (_Float16)(accV[ct][lt][i] + bvS[c]);
      }
}

// ---------------------------------------------------------------------------
// Kernel 2: energy = Q K^T + row softmax. One WG = (b, 32 l rows).
// Wave w owns columns m in [128w, 128w+128): 2 l-tiles x 8 m-tiles.
// ---------------------------------------------------------------------------
__global__ __launch_bounds__(256) void attn_kernel(
    const _Float16* __restrict__ Qh, const _Float16* __restrict__ Kh,
    float* __restrict__ att) {
  const int b  = blockIdx.x >> 5;
  const int l0 = (blockIdx.x & 31) * 32;
  const int tid = threadIdx.x;
  const int w = tid >> 5, lane = tid & 31, g = lane >> 4, n = lane & 15;

  __shared__ _Float16 Qt[32 * 72];   // padded stride 72 (conflict-free)
  __shared__ float red[32][8];
  __shared__ float rowstat[32];

  {  // stage Q block [32 l][64 d]
    const _Float16* src = Qh + (size_t)(b * 1024 + l0) * 64;
    const int l = tid >> 3, d0 = (tid & 7) * 8;
    *(v8h*)(Qt + l * 72 + d0) = *(const v8h*)(src + l * 64 + d0);
  }
  __syncthreads();

  v8f acc[2][8] = {};
#pragma unroll
  for (int kk = 0; kk < 2; ++kk) {          // K loop over d=64
    v16h a0 = load_a_f16(Qt + kk * 32, 72);
    v16h a1 = load_a_f16(Qt + 16 * 72 + kk * 32, 72);
#pragma unroll
    for (int mt = 0; mt < 8; ++mt) {
      const _Float16* kb = Kh + (size_t)(b * 1024 + w * 128 + mt * 16) * 64 + kk * 32;
      v16h bf = load_b_f16(kb, 64);
      acc[0][mt] = wmma_f16(a0, bf, acc[0][mt]);
      acc[1][mt] = wmma_f16(a1, bf, acc[1][mt]);
    }
  }

  // ---- row max (over all m) ----
#pragma unroll
  for (int lt = 0; lt < 2; ++lt)
#pragma unroll
    for (int i = 0; i < 8; ++i) {
      float m = acc[lt][0][i];
#pragma unroll
      for (int mt = 1; mt < 8; ++mt) m = fmaxf(m, acc[lt][mt][i]);
#pragma unroll
      for (int off = 1; off <= 8; off <<= 1) m = fmaxf(m, __shfl_xor(m, off, 32));
      if (n == 0) red[lt * 16 + i + 8 * g][w] = m;
    }
  __syncthreads();
  if (tid < 32) {
    float m = red[tid][0];
#pragma unroll
    for (int j = 1; j < 8; ++j) m = fmaxf(m, red[tid][j]);
    rowstat[tid] = m;
  }
  __syncthreads();

  // ---- exp and row sum ----
#pragma unroll
  for (int lt = 0; lt < 2; ++lt)
#pragma unroll
    for (int i = 0; i < 8; ++i) {
      const float mx = rowstat[lt * 16 + i + 8 * g];
      float s = 0.f;
#pragma unroll
      for (int mt = 0; mt < 8; ++mt) {
        const float p = __expf(acc[lt][mt][i] - mx);
        acc[lt][mt][i] = p;
        s += p;
      }
#pragma unroll
      for (int off = 1; off <= 8; off <<= 1) s += __shfl_xor(s, off, 32);
      if (n == 0) red[lt * 16 + i + 8 * g][w] = s;
    }
  __syncthreads();
  if (tid < 32) {
    float s = red[tid][0];
#pragma unroll
    for (int j = 1; j < 8; ++j) s += red[tid][j];
    rowstat[tid] = s;
  }
  __syncthreads();

  // ---- normalize and store fp32 attention ----
#pragma unroll
  for (int lt = 0; lt < 2; ++lt)
#pragma unroll
    for (int i = 0; i < 8; ++i) {
      const int row = lt * 16 + i + 8 * g;
      const float rinv = 1.0f / rowstat[row];
      float* dst = att + ((size_t)(b * 1024 + l0 + row)) * 1024 + w * 128 + n;
#pragma unroll
      for (int mt = 0; mt < 8; ++mt) dst[mt * 16] = acc[lt][mt][i] * rinv;
    }
}

// ---------------------------------------------------------------------------
// Kernel 3: out = gamma * (V @ att^T) + x. One WG = (b, 128 c, 64 l).
// A = Vh[c][m] row-major f16; B = att[l][m] fp32 (column-major-[N][K] for WMMA).
// ---------------------------------------------------------------------------
__global__ __launch_bounds__(256) void out_kernel(
    const _Float16* __restrict__ Vh, const float* __restrict__ att,
    const float* __restrict__ x,     const float* __restrict__ gamma,
    float* __restrict__ out) {
  const int blk = blockIdx.x;
  const int b = blk >> 6;
  const int r = blk & 63;
  const int cblk = (r >> 4) * 128;
  const int lblk = (r & 15) * 64;
  const int tid = threadIdx.x;
  const int w = tid >> 5, lane = tid & 31, g = lane >> 4, n = lane & 15;

  v8f acc[4] = {};
  for (int kk = 0; kk < 32; ++kk) {  // K loop over m=1024
    v16h a = load_a_f16(Vh + (size_t)(b * 512 + cblk + w * 16) * 1024 + kk * 32, 1024);
#pragma unroll
    for (int lt = 0; lt < 4; ++lt) {
      v16h bf = load_b_f32(att + (size_t)(b * 1024 + lblk + lt * 16) * 1024 + kk * 32, 1024);
      acc[lt] = wmma_f16(a, bf, acc[lt]);
    }
  }

  const float gm = gamma[0];
#pragma unroll
  for (int lt = 0; lt < 4; ++lt)
#pragma unroll
    for (int i = 0; i < 8; ++i) {
      const int c = cblk + w * 16 + i + 8 * g;
      const int l = lblk + lt * 16 + n;
      const size_t idx = (size_t)(b * 512 + c) * 1024 + l;
      out[idx] = gm * acc[lt][i] + x[idx];
    }
}

extern "C" void kernel_launch(void* const* d_in, const int* in_sizes, int n_in,
                              void* d_out, int out_size, void* d_ws, size_t ws_size,
                              hipStream_t stream) {
  const float* x     = (const float*)d_in[0];
  const float* Wq    = (const float*)d_in[1];
  const float* bq    = (const float*)d_in[2];
  const float* Wk    = (const float*)d_in[3];
  const float* bk    = (const float*)d_in[4];
  const float* Wv    = (const float*)d_in[5];
  const float* bv    = (const float*)d_in[6];
  const float* gamma = (const float*)d_in[7];

  float* out = (float*)d_out;
  float* att = out + (size_t)32 * 512 * 1024;   // attention region of d_out

  _Float16* Qh = (_Float16*)d_ws;               // [32][1024][64]
  _Float16* Kh = Qh + (size_t)32 * 1024 * 64;   // [32][1024][64]
  _Float16* Vh = Kh + (size_t)32 * 1024 * 64;   // [32][512][1024]  (40MB total)

  proj_kernel<<<1024, 256, 0, stream>>>(x, Wq, bq, Wk, bk, Wv, bv, Qh, Kh, Vh);
  attn_kernel<<<1024, 256, 0, stream>>>(Qh, Kh, att);
  out_kernel<<<2048, 256, 0, stream>>>(Vh, att, x, gamma, out);
}
